// GraphEncoder_69724499083755
// MI455X (gfx1250) — matrix operand order repaired and checked by reading, not compiled, not run on hardware
//
#include <hip/hip_runtime.h>
#include <math.h>

// ---------------- problem constants (from reference) ----------------
constexpr int NN = 50000;   // nodes
constexpr int NE = 400000;  // edges (self loops appended logically)
constexpr int NB = 512;     // graphs in batch

// ---------------- WMMA types ----------------
typedef __attribute__((ext_vector_type(16))) __bf16 v16bf;
typedef __attribute__((ext_vector_type(8)))  float  v8f;

__device__ __forceinline__ unsigned short bf16bits(float f) {
  unsigned u = __builtin_bit_cast(unsigned, f);
  u = (u + 0x7FFFu + ((u >> 16) & 1u)) >> 16;   // round-to-nearest-even
  return (unsigned short)u;
}

__device__ __forceinline__ void atomicMaxFloat(float* addr, float v) {
  if (v >= 0.0f) atomicMax((int*)addr, __float_as_int(v));
  else           atomicMin((unsigned int*)addr, __float_as_uint(v));
}

// ---------------- fragment packing ----------------
// A (f32 row-major [M,K]) -> bf16 fragments. Fragment = 32 lanes x 16 elems,
// lane-contiguous (32B/lane). Fragment order: mtile-major, then kchunk.
// A-fragment lane layout used here (contiguous k order, not the ISA 7.12.2 table order):
//   lane = (K-half)*16 + M%16 ; elem j=0..15 -> K=khalf*16+j
__global__ void pack_a(const float* __restrict__ A, unsigned short* __restrict__ P,
                       int M, int K) {
  int t = blockIdx.x * blockDim.x + threadIdx.x;            // frag*32 + lane
  int kchunks = K >> 5;
  int nfrag = (M >> 4) * kchunks;
  if (t >= nfrag * 32) return;
  int lane = t & 31, frag = t >> 5;
  int mtile = frag / kchunks, kc = frag % kchunks;
  int m  = (mtile << 4) + (lane & 15);
  int k0 = (kc << 5) + (lane >> 4) * 16;
  const float* pa = A + (size_t)m * K + k0;
  unsigned short* out = P + (size_t)t * 16;
#pragma unroll
  for (int j = 0; j < 8; ++j) out[j] = bf16bits(pa[j]);
#pragma unroll
  for (int j = 0; j < 8; ++j) out[8 + j] = bf16bits(pa[8 + j]);
}

// B (f32 row-major [K,N]) -> bf16 fragments. Fragment order: kchunk-major? No:
// frag = kc * (N/16) + ntile.  B-fragment lane layout (32x16 bf16):
//   lane = khalf*16 + N%16 ; elem j -> K = khalf*16 + j
__global__ void pack_b(const float* __restrict__ B, unsigned short* __restrict__ P,
                       int K, int N) {
  int t = blockIdx.x * blockDim.x + threadIdx.x;
  int tilesN = N >> 4;
  int nfrag = (K >> 5) * tilesN;
  if (t >= nfrag * 32) return;
  int lane = t & 31, frag = t >> 5;
  int kc = frag / tilesN, nt = frag % tilesN;
  int n = (nt << 4) + (lane & 15);
  int kbase = (kc << 5) + (lane >> 4) * 16;
  unsigned short* out = P + (size_t)t * 16;
#pragma unroll
  for (int j = 0; j < 16; ++j) out[j] = bf16bits(B[(size_t)(kbase + j) * N + n]);
}

// ---------------- packed bf16 WMMA GEMM ----------------
// C[M,N] = act(A @ B + bias). One wave per 16x64 C strip: 1 A-frag load feeds
// 4 WMMAs. Requires M%16==0, K%32==0, N%64==0 (true for all calls).
__global__ __launch_bounds__(256)
void wmma_gemm_pk(const unsigned short* __restrict__ Apk,
                  const unsigned short* __restrict__ Bpk,
                  const float* __restrict__ bias, float* __restrict__ C,
                  int M, int K, int Nout, float slope) {
  const int lane = threadIdx.x & 31;
  const int wid  = blockIdx.x * (blockDim.x >> 5) + (threadIdx.x >> 5);
  const int tilesN  = Nout >> 4;
  const int strips  = tilesN >> 2;                 // 16x64 strips per row band
  const int total   = (M >> 4) * strips;
  if (wid >= total) return;                        // wave-uniform exit
  const int mtile = wid / strips;
  const int nstrip = wid % strips;
  const int kchunks = K >> 5;

  const v16bf* af = (const v16bf*)Apk + (size_t)(mtile * kchunks) * 32 + lane;
  const v16bf* bf = (const v16bf*)Bpk + (size_t)(nstrip * 4) * 32 + lane;
  const size_t bstride = (size_t)tilesN * 32;      // v16bf units per kchunk

  v8f acc0 = {}, acc1 = {}, acc2 = {}, acc3 = {};
  for (int kc = 0; kc < kchunks; ++kc) {
    v16bf a  = af[(size_t)kc * 32];
    v16bf b0 = bf[(size_t)kc * bstride];
    v16bf b1 = bf[(size_t)kc * bstride + 32];
    v16bf b2 = bf[(size_t)kc * bstride + 64];
    v16bf b3 = bf[(size_t)kc * bstride + 96];
    acc0 = __builtin_amdgcn_wmma_f32_16x16x32_bf16(false, a, false, b0, (short)0, acc0, false, false);
    acc1 = __builtin_amdgcn_wmma_f32_16x16x32_bf16(false, a, false, b1, (short)0, acc1, false, false);
    acc2 = __builtin_amdgcn_wmma_f32_16x16x32_bf16(false, a, false, b2, (short)0, acc2, false, false);
    acc3 = __builtin_amdgcn_wmma_f32_16x16x32_bf16(false, a, false, b3, (short)0, acc3, false, false);
  }

  // C layout: VGPR r, lanes 0-15 -> M=r, lanes 16-31 -> M=r+8
  const int khalf = lane >> 4;
  const int row0  = (mtile << 4) + khalf * 8;
  v8f accs[4] = {acc0, acc1, acc2, acc3};
#pragma unroll
  for (int sub = 0; sub < 4; ++sub) {
    const int ncol = (nstrip << 6) + (sub << 4) + (lane & 15);
    const float bv = bias ? bias[ncol] : 0.0f;
    float* crow = C + (size_t)row0 * Nout + ncol;
#pragma unroll
    for (int r = 0; r < 8; ++r) {
      float v = accs[sub][r] + bv;
      if (slope >= 0.0f) v = v > 0.0f ? v : v * slope;
      crow[(size_t)r * Nout] = v;
    }
  }
}

// ---------------- helpers ----------------
__global__ void fill_f32(float* __restrict__ p, float v, long n) {
  long t = (long)blockIdx.x * blockDim.x + threadIdx.x;
  if (t < n) p[t] = v;
}

__global__ void gather_embed(const int* __restrict__ x, const float* __restrict__ emb,
                             float* __restrict__ h0, int n) {
  int t = blockIdx.x * blockDim.x + threadIdx.x;
  if (t >= n * 32) return;
  int node = t >> 5, c = t & 31;
  int tok = x[node];
  ((float4*)(h0 + (size_t)node * 128))[c] = ((const float4*)(emb + (size_t)tok * 128))[c];
}

__global__ void alpha_kernel(const float* __restrict__ Hw, const float* __restrict__ a_src,
                             const float* __restrict__ a_dst, float* __restrict__ asb,
                             float* __restrict__ adb, int n) {
  int wid = blockIdx.x * (blockDim.x >> 5) + (threadIdx.x >> 5);
  int lane = threadIdx.x & 31;
  if (wid >= n) return;
  const float* row = Hw + (size_t)wid * 256;
#pragma unroll
  for (int h = 0; h < 2; ++h) {
    float ss = 0.0f, sd = 0.0f;
    for (int i = lane; i < 128; i += 32) {
      float v = row[h * 128 + i];
      ss += v * a_src[h * 128 + i];
      sd += v * a_dst[h * 128 + i];
    }
    for (int o = 16; o > 0; o >>= 1) {
      ss += __shfl_down(ss, o, 32);
      sd += __shfl_down(sd, o, 32);
    }
    if (lane == 0) { asb[wid * 2 + h] = ss; adb[wid * 2 + h] = sd; }
  }
}

__device__ __forceinline__ void edge_sd(const int* ei, int e, int nE, int& s, int& d) {
  if (e < nE) { s = ei[e]; d = ei[nE + e]; } else { s = d = e - nE; }
}

__global__ void edge_score_max(const int* __restrict__ ei, const float* __restrict__ asb,
                               const float* __restrict__ adb, float* __restrict__ sc,
                               float* __restrict__ mb, int nE, int n) {
  long t = (long)blockIdx.x * blockDim.x + threadIdx.x;
  if (t >= (long)(nE + n) * 2) return;
  int e = (int)(t >> 1), h = (int)(t & 1);
  int s, d; edge_sd(ei, e, nE, s, d);
  float v = asb[s * 2 + h] + adb[d * 2 + h];
  v = v > 0.0f ? v : v * 0.2f;
  sc[t] = v;
  atomicMaxFloat(&mb[d * 2 + h], v);
}

__global__ void edge_exp_sum(const int* __restrict__ ei, const float* __restrict__ sc,
                             const float* __restrict__ mb, float* __restrict__ ex,
                             float* __restrict__ den, int nE, int n) {
  long t = (long)blockIdx.x * blockDim.x + threadIdx.x;
  if (t >= (long)(nE + n) * 2) return;
  int e = (int)(t >> 1), h = (int)(t & 1);
  int s, d; edge_sd(ei, e, nE, s, d); (void)s;
  float v = __expf(sc[t] - mb[d * 2 + h]);
  ex[t] = v;
  atomicAdd(&den[d * 2 + h], v);
}

__global__ void edge_aggregate(const int* __restrict__ ei, const float* __restrict__ Hw,
                               const float* __restrict__ ex, const float* __restrict__ den,
                               float* __restrict__ agg, int nE, int n) {
  int wid = blockIdx.x * (blockDim.x >> 5) + (threadIdx.x >> 5);
  int lane = threadIdx.x & 31;
  if (wid >= nE + n) return;
  int s, d; edge_sd(ei, wid, nE, s, d);
  float a0 = ex[(size_t)wid * 2 + 0] / den[d * 2 + 0];
  float a1 = ex[(size_t)wid * 2 + 1] / den[d * 2 + 1];
  const float* hs = Hw + (size_t)s * 256;
  float* od = agg + (size_t)d * 256;
#pragma unroll
  for (int i = lane; i < 256; i += 32) {
    float w = (i < 128) ? a0 : a1;
    atomicAdd(&od[i], hs[i] * w);
  }
}

__global__ void bias_act(const float* __restrict__ in, const float* __restrict__ bias,
                         float* __restrict__ out, long total, int cols, float slope) {
  long t = (long)blockIdx.x * blockDim.x + threadIdx.x;
  if (t >= total) return;
  int c = (int)(t % cols);
  float v = in[t] + bias[c];
  if (slope >= 0.0f) v = v > 0.0f ? v : v * slope;
  out[t] = v;
}

__global__ void gate_kernel(const float* __restrict__ g1, const float* __restrict__ g_w2,
                            const float* __restrict__ g_b2, float* __restrict__ gate, int n) {
  int wid = blockIdx.x * (blockDim.x >> 5) + (threadIdx.x >> 5);
  int lane = threadIdx.x & 31;
  if (wid >= n) return;
  const float* row = g1 + (size_t)wid * 128;
  float s = 0.0f;
  for (int i = lane; i < 128; i += 32) s += row[i] * g_w2[i];
  for (int o = 16; o > 0; o >>= 1) s += __shfl_down(s, o, 32);
  if (lane == 0) gate[wid] = s + g_b2[0];
}

__global__ void gmax_kernel(const int* __restrict__ batch, const float* __restrict__ gate,
                            float* __restrict__ gm, int n) {
  int t = blockIdx.x * blockDim.x + threadIdx.x;
  if (t >= n) return;
  atomicMaxFloat(&gm[batch[t]], gate[t]);
}

__global__ void gexp_kernel(const int* __restrict__ batch, const float* __restrict__ gate,
                            const float* __restrict__ gm, float* __restrict__ ge,
                            float* __restrict__ gs, int n) {
  int t = blockIdx.x * blockDim.x + threadIdx.x;
  if (t >= n) return;
  int b = batch[t];
  float v = __expf(gate[t] - gm[b]);
  ge[t] = v;
  atomicAdd(&gs[b], v);
}

__global__ void pool_kernel(const int* __restrict__ batch, const float* __restrict__ oc,
                            const float* __restrict__ ge, const float* __restrict__ gs,
                            float* __restrict__ pooled, int n) {
  int wid = blockIdx.x * (blockDim.x >> 5) + (threadIdx.x >> 5);
  int lane = threadIdx.x & 31;
  if (wid >= n) return;
  int b = batch[wid];
  float w = ge[wid] / gs[b];
  const float* row = oc + (size_t)wid * 256;
  float* pr = pooled + (size_t)b * 256;
#pragma unroll
  for (int i = lane; i < 256; i += 32) atomicAdd(&pr[i], row[i] * w);
}

__global__ void layernorm_kernel(const float* __restrict__ pooled, const float* __restrict__ g,
                                 const float* __restrict__ b, float* __restrict__ xn, int rows) {
  int wid = blockIdx.x * (blockDim.x >> 5) + (threadIdx.x >> 5);
  int lane = threadIdx.x & 31;
  if (wid >= rows) return;
  const float* row = pooled + (size_t)wid * 256;
  float s = 0.0f, s2 = 0.0f;
  for (int i = lane; i < 256; i += 32) { float v = row[i]; s += v; s2 += v * v; }
  for (int o = 16; o > 0; o >>= 1) { s += __shfl_down(s, o, 32); s2 += __shfl_down(s2, o, 32); }
  s  = __shfl(s, 0, 32);
  s2 = __shfl(s2, 0, 32);
  float mu  = s * (1.0f / 256.0f);
  float var = s2 * (1.0f / 256.0f) - mu * mu;
  float rstd = rsqrtf(var + 1e-5f);
  float* orow = xn + (size_t)wid * 256;
  for (int i = lane; i < 256; i += 32) orow[i] = (row[i] - mu) * rstd * g[i] + b[i];
}

// ---------------- host-side orchestration ----------------
extern "C" void kernel_launch(void* const* d_in, const int* in_sizes, int n_in,
                              void* d_out, int out_size, void* d_ws, size_t ws_size,
                              hipStream_t stream) {
  const int*   x      = (const int*)  d_in[0];
  const int*   ei     = (const int*)  d_in[1];
  const int*   batch  = (const int*)  d_in[2];
  const float* emb    = (const float*)d_in[3];
  const float* W1     = (const float*)d_in[4];
  const float* a_src1 = (const float*)d_in[5];
  const float* a_dst1 = (const float*)d_in[6];
  const float* b1     = (const float*)d_in[7];
  const float* W2     = (const float*)d_in[8];
  const float* a_src2 = (const float*)d_in[9];
  const float* a_dst2 = (const float*)d_in[10];
  const float* b2     = (const float*)d_in[11];
  const float* g_w1   = (const float*)d_in[12];
  const float* g_b1   = (const float*)d_in[13];
  const float* g_w2   = (const float*)d_in[14];
  const float* g_b2   = (const float*)d_in[15];
  const float* ln_g   = (const float*)d_in[16];
  const float* ln_b   = (const float*)d_in[17];
  const float* f_w1   = (const float*)d_in[18];
  const float* f_b1   = (const float*)d_in[19];
  const float* f_w2   = (const float*)d_in[20];
  const float* f_b2   = (const float*)d_in[21];

  float* out_conv = (float*)d_out;                       // N x 256 (also layer-1 scratch)
  float* hidden   = (float*)d_out + (size_t)NN * 256;    // B x 128

  float* ws = (float*)d_ws;
  size_t off = 0;
  auto alloc = [&](size_t nf) { float* p = ws + off; off += (nf + 255) & ~(size_t)255; return p; };
  float* h0   = alloc((size_t)NN * 128);   // embeddings; reused later as g1
  float* Hw   = alloc((size_t)NN * 256);   // h @ W (per layer)
  unsigned short* Apk  = (unsigned short*)alloc((size_t)NN * 128);  // N*256 bf16
  unsigned short* W1pk = (unsigned short*)alloc(16384);
  unsigned short* W2pk = (unsigned short*)alloc(32768);
  unsigned short* Gw1p = (unsigned short*)alloc(16384);
  unsigned short* Fw1p = (unsigned short*)alloc(32768);
  unsigned short* Fw2p = (unsigned short*)alloc(16384);
  float* asb  = alloc((size_t)NN * 2);
  float* adb  = alloc((size_t)NN * 2);
  float* mb   = alloc((size_t)NN * 2);
  float* den  = alloc((size_t)NN * 2);
  float* sc   = alloc((size_t)(NE + NN) * 2);
  float* ex   = alloc((size_t)(NE + NN) * 2);
  float* gate = alloc(NN);
  float* ge   = alloc(NN);
  float* gm   = alloc(NB);
  float* gs   = alloc(NB);
  float* pooled = alloc((size_t)NB * 256);
  float* xn     = alloc((size_t)NB * 256);
  float* f1     = alloc((size_t)NB * 256);

  const int TPB = 256;                                 // 8 waves/block (wave32)
  auto tgrid = [](long n) { return dim3((unsigned)((n + 255) / 256)); };
  auto wgrid = [](long w) { return dim3((unsigned)((w + 7) / 8)); };
  const float NEG_INF = -__builtin_inff();

  // GEMM = pack A to bf16 fragments, then 16x64-strip WMMA kernel
  auto gemm = [&](const float* Ain, int M, int K, const unsigned short* Bpk,
                  const float* bias, float* Cout, int Nout, float slope) {
    pack_a<<<tgrid((long)(M / 16) * (K / 32) * 32), TPB, 0, stream>>>(Ain, Apk, M, K);
    wmma_gemm_pk<<<wgrid((long)(M / 16) * (Nout / 64)), TPB, 0, stream>>>(
        Apk, Bpk, bias, Cout, M, K, Nout, slope);
  };

  // pack all weights once
  pack_b<<<tgrid((long)(128/32)*(256/16)*32), TPB, 0, stream>>>(W1,   W1pk, 128, 256);
  pack_b<<<tgrid((long)(256/32)*(256/16)*32), TPB, 0, stream>>>(W2,   W2pk, 256, 256);
  pack_b<<<tgrid((long)(256/32)*(128/16)*32), TPB, 0, stream>>>(g_w1, Gw1p, 256, 128);
  pack_b<<<tgrid((long)(256/32)*(256/16)*32), TPB, 0, stream>>>(f_w1, Fw1p, 256, 256);
  pack_b<<<tgrid((long)(256/32)*(128/16)*32), TPB, 0, stream>>>(f_w2, Fw2p, 256, 128);

  // 0) embedding gather
  gather_embed<<<tgrid((long)NN * 32), TPB, 0, stream>>>(x, emb, h0, NN);

  // ---- GAT layer 1 (h1 lands in out_conv region) ----
  gemm(h0, NN, 128, W1pk, nullptr, Hw, 256, -1.0f);
  alpha_kernel<<<wgrid(NN), TPB, 0, stream>>>(Hw, a_src1, a_dst1, asb, adb, NN);
  fill_f32<<<tgrid((long)NN * 2), TPB, 0, stream>>>(mb, NEG_INF, (long)NN * 2);
  fill_f32<<<tgrid((long)NN * 2), TPB, 0, stream>>>(den, 0.0f, (long)NN * 2);
  fill_f32<<<tgrid((long)NN * 256), TPB, 0, stream>>>(out_conv, 0.0f, (long)NN * 256);
  edge_score_max<<<tgrid((long)(NE + NN) * 2), TPB, 0, stream>>>(ei, asb, adb, sc, mb, NE, NN);
  edge_exp_sum  <<<tgrid((long)(NE + NN) * 2), TPB, 0, stream>>>(ei, sc, mb, ex, den, NE, NN);
  edge_aggregate<<<wgrid(NE + NN), TPB, 0, stream>>>(ei, Hw, ex, den, out_conv, NE, NN);
  bias_act<<<tgrid((long)NN * 256), TPB, 0, stream>>>(out_conv, b1, out_conv, (long)NN * 256, 256, 0.05f);

  // ---- GAT layer 2 ----
  gemm(out_conv, NN, 256, W2pk, nullptr, Hw, 256, -1.0f);
  alpha_kernel<<<wgrid(NN), TPB, 0, stream>>>(Hw, a_src2, a_dst2, asb, adb, NN);
  fill_f32<<<tgrid((long)NN * 2), TPB, 0, stream>>>(mb, NEG_INF, (long)NN * 2);
  fill_f32<<<tgrid((long)NN * 2), TPB, 0, stream>>>(den, 0.0f, (long)NN * 2);
  fill_f32<<<tgrid((long)NN * 256), TPB, 0, stream>>>(out_conv, 0.0f, (long)NN * 256);
  edge_score_max<<<tgrid((long)(NE + NN) * 2), TPB, 0, stream>>>(ei, asb, adb, sc, mb, NE, NN);
  edge_exp_sum  <<<tgrid((long)(NE + NN) * 2), TPB, 0, stream>>>(ei, sc, mb, ex, den, NE, NN);
  edge_aggregate<<<wgrid(NE + NN), TPB, 0, stream>>>(ei, Hw, ex, den, out_conv, NE, NN);
  bias_act<<<tgrid((long)NN * 256), TPB, 0, stream>>>(out_conv, b2, out_conv, (long)NN * 256, 256, -1.0f);

  // ---- gating + softmax pooling over batch ----
  float* g1 = h0;  // reuse N x 128
  gemm(out_conv, NN, 256, Gw1p, g_b1, g1, 128, 0.05f);
  gate_kernel<<<wgrid(NN), TPB, 0, stream>>>(g1, g_w2, g_b2, gate, NN);
  fill_f32<<<tgrid(NB), TPB, 0, stream>>>(gm, NEG_INF, NB);
  fill_f32<<<tgrid(NB), TPB, 0, stream>>>(gs, 0.0f, NB);
  fill_f32<<<tgrid((long)NB * 256), TPB, 0, stream>>>(pooled, 0.0f, (long)NB * 256);
  gmax_kernel<<<tgrid(NN), TPB, 0, stream>>>(batch, gate, gm, NN);
  gexp_kernel<<<tgrid(NN), TPB, 0, stream>>>(batch, gate, gm, ge, gs, NN);
  pool_kernel<<<wgrid(NN), TPB, 0, stream>>>(batch, out_conv, ge, gs, pooled, NN);

  // ---- LayerNorm + MLP head ----
  layernorm_kernel<<<wgrid(NB), TPB, 0, stream>>>(pooled, ln_g, ln_b, xn, NB);
  gemm(xn, NB, 256, Fw1p, f_b1, f1, 256, 0.05f);
  gemm(f1, NB, 256, Fw2p, f_b2, hidden, 128, -1.0f);
}
